// GATLayer_704374636775
// MI455X (gfx1250) — matrix operand order repaired and checked by reading, not compiled, not run on hardware
//
#include <hip/hip_runtime.h>
#include <hip/hip_bf16.h>

typedef __attribute__((ext_vector_type(2)))  float    v2f;
typedef __attribute__((ext_vector_type(8)))  float    v8f;
typedef __attribute__((ext_vector_type(16))) _Float16 v16h;

#define NN    4096
#define FIN   256
#define FOUT  64
#define NH    8
#define ALPHA 0.2f
#define L2E   1.44269504088896340736f

// workspace layout (bytes)
#define OFF_HT    0u                       // 8*64*4096*2  = 4 MB  (f16, transposed h)
#define OFF_F1    4194304u                 // 8*4096*4     = 128 KB
#define OFF_F2    4325376u                 // 128 KB
#define OFF_MASK  4456448u                 // 4096*128*4   = 2 MB  (adj bitmask)
#define OFF_STATS 6553600u                 // 8*4096*2*4   = 256 KB (m_i, 1/denom_i)

// ---------------------------------------------------------------------------
// Kernel 1: h = x @ W  (per head), stored transposed as f16; also f1,f2.
// One block = (head, 16-row tile). 4 waves, each owns a 16-feature tile.
// Exact fp32 via v_wmma_f32_16x16x4_f32, K-loop of 64 steps.
// ---------------------------------------------------------------------------
__global__ void __launch_bounds__(128)
k1_proj(const float* __restrict__ x, const float* __restrict__ W,
        const float* __restrict__ a, _Float16* __restrict__ hT,
        float* __restrict__ f1, float* __restrict__ f2)
{
    __shared__ float hs[16][66];
    const int hd = blockIdx.x & 7, nt = blockIdx.x >> 3;
    const int tid = threadIdx.x, wave = tid >> 5, lane = tid & 31;
    const int M = lane & 15;              // A-row / D-col index
    const int kb = (lane >> 4) * 2;       // fp32 A: lanes16-31 hold K+2,K+3
    const int obase = wave * 16, ncol = lane & 15;

    const float* xrow = x + (size_t)(nt * 16 + M) * FIN;
    const float* Wh   = W + (size_t)hd * FIN * FOUT;

    v8f acc = {};
    for (int k0 = 0; k0 < FIN; k0 += 4) {
        v2f av, bv;
        av.x = xrow[k0 + kb];
        av.y = xrow[k0 + kb + 1];
        bv.x = Wh[(k0 + kb)     * FOUT + obase + ncol];
        bv.y = Wh[(k0 + kb + 1) * FOUT + obase + ncol];
        acc = __builtin_amdgcn_wmma_f32_16x16x4_f32(
            false, av, false, bv, (short)0, acc, false, false);
    }

    const int rbase = (lane >> 4) * 8;    // D: lanes16-31 hold rows M+8
    #pragma unroll
    for (int r = 0; r < 8; ++r) {
        const int m = rbase + r;
        const float v = acc[r];
        hs[m][obase + ncol] = v;
        hT[(size_t)(hd * FOUT + obase + ncol) * NN + nt * 16 + m] = (_Float16)v;
    }
    __syncthreads();

    if (tid < 16) {
        float s1 = 0.f, s2 = 0.f;
        #pragma unroll 8
        for (int o = 0; o < FOUT; ++o) {
            const float hv = hs[tid][o];
            s1 += hv * a[hd * 128 + o];
            s2 += hv * a[hd * 128 + 64 + o];
        }
        f1[hd * NN + nt * 16 + tid] = s1;
        f2[hd * NN + nt * 16 + tid] = s2;
    }
}

// ---------------------------------------------------------------------------
// Kernel 2: per row i — pack adj row into bitmask (b128 streaming loads);
// per head: m = max_{connected j} leaky(f1_i + f2_j), denom = sum exp(z-m).
// One block = one row, 256 threads; each thread owns 4x4 contiguous j.
// ---------------------------------------------------------------------------
__global__ void __launch_bounds__(256)
k2_stats(const int* __restrict__ adj, const float* __restrict__ f1,
         const float* __restrict__ f2, unsigned int* __restrict__ amask,
         float* __restrict__ rstats)
{
    __shared__ unsigned int mw[128];
    __shared__ float red[8 * 256];
    __shared__ float mfin[8];

    const int i = blockIdx.x, tid = threadIdx.x;
    const int g = tid >> 5, l = tid & 31;

    for (int k = tid; k < 128; k += 256) mw[k] = 0u;
    __syncthreads();

    // stream adjacency row: 4 iterations of int4 per thread
    unsigned int bits = 0u;                       // 16 adjacency bits: it*4+c
    #pragma unroll
    for (int it = 0; it < 4; ++it) {
        const int j0 = it * 1024 + tid * 4;
        const int4 av = *(const int4*)(adj + (size_t)i * NN + j0);
        unsigned int nib = 0u;
        if (av.x > 0) nib |= 1u;
        if (av.y > 0) nib |= 2u;
        if (av.z > 0) nib |= 4u;
        if (av.w > 0) nib |= 8u;
        if (nib) atomicOr(&mw[j0 >> 5], nib << ((tid & 7) << 2));
        bits |= nib << (it * 4);
    }
    __syncthreads();
    for (int k = tid; k < 128; k += 256) amask[(size_t)i * 128 + k] = mw[k];

    float f1r[NH];
    #pragma unroll
    for (int h = 0; h < NH; ++h) f1r[h] = f1[h * NN + i];

    // pass 1: per-head max of f2 over connected j (leaky is monotone)
    float ml[NH];
    #pragma unroll
    for (int h = 0; h < NH; ++h) ml[h] = -3.0e38f;
    #pragma unroll
    for (int it = 0; it < 4; ++it) {
        const int j0 = it * 1024 + tid * 4;
        const unsigned int nib = (bits >> (it * 4)) & 15u;
        if (nib) {
            #pragma unroll
            for (int h = 0; h < NH; ++h) {
                const float4 fv = *(const float4*)(f2 + h * NN + j0);
                if (nib & 1u) ml[h] = fmaxf(ml[h], fv.x);
                if (nib & 2u) ml[h] = fmaxf(ml[h], fv.y);
                if (nib & 4u) ml[h] = fmaxf(ml[h], fv.z);
                if (nib & 8u) ml[h] = fmaxf(ml[h], fv.w);
            }
        }
    }
    #pragma unroll
    for (int h = 0; h < NH; ++h) red[h * 256 + tid] = ml[h];
    __syncthreads();
    {
        float v = -3.0e38f;
        for (int t = l; t < 256; t += 32) v = fmaxf(v, red[g * 256 + t]);
        #pragma unroll
        for (int off = 16; off > 0; off >>= 1)
            v = fmaxf(v, __shfl_xor(v, off, 32));
        if (l == 0) mfin[g] = v;
    }
    __syncthreads();

    float mz[NH];
    #pragma unroll
    for (int h = 0; h < NH; ++h) {
        const float t = f1r[h] + mfin[h];
        mz[h] = fmaxf(t, ALPHA * t);              // max logit of the row
    }

    // pass 2: softmax denominator
    float sl[NH];
    #pragma unroll
    for (int h = 0; h < NH; ++h) sl[h] = 0.f;
    #pragma unroll
    for (int it = 0; it < 4; ++it) {
        const int j0 = it * 1024 + tid * 4;
        const unsigned int nib = (bits >> (it * 4)) & 15u;
        if (nib) {
            #pragma unroll
            for (int h = 0; h < NH; ++h) {
                const float4 fv = *(const float4*)(f2 + h * NN + j0);
                const float fvv[4] = {fv.x, fv.y, fv.z, fv.w};
                #pragma unroll
                for (int c = 0; c < 4; ++c) {
                    if (nib & (1u << c)) {
                        float z = f1r[h] + fvv[c];
                        z = fmaxf(z, ALPHA * z);
                        sl[h] += exp2f(fmaf(z, L2E, -mz[h] * L2E));
                    }
                }
            }
        }
    }
    __syncthreads();
    #pragma unroll
    for (int h = 0; h < NH; ++h) red[h * 256 + tid] = sl[h];
    __syncthreads();
    {
        float v = 0.f;
        for (int t = l; t < 256; t += 32) v += red[g * 256 + t];
        #pragma unroll
        for (int off = 16; off > 0; off >>= 1)
            v += __shfl_xor(v, off, 32);
        if (l == 0) {
            rstats[(size_t)(g * NN + i) * 2 + 0] = mz[g];
            rstats[(size_t)(g * NN + i) * 2 + 1] = 1.0f / v;
        }
    }
}

// ---------------------------------------------------------------------------
// Kernel 3: fused O = softmax(e) @ h using v_wmma_f32_16x16x32_f16.
// One block = (head, 16-query-row tile); 4 waves each own a 1024-wide j-range
// and all 64 features (4 f32 accumulators). P tile built per-lane directly in
// a contiguous-k A layout (k = 16*(lane/16) + s); B fragments are contiguous 32 B loads from hT.
// Mask bits and f2 are LDS-resident for the whole block.
// ---------------------------------------------------------------------------
__global__ void __launch_bounds__(128)
k3_attn(const _Float16* __restrict__ hT, const float* __restrict__ f1,
        const float* __restrict__ f2, const unsigned int* __restrict__ amask,
        const float* __restrict__ rstats, float* __restrict__ out)
{
    __shared__ float f2s[NN];                 // 16 KB: this head's f2
    __shared__ unsigned int msk[16 * 128];    //  8 KB: 16 rows of adj bitmask
    __shared__ float redo[4 * 4 * 8 * 32];    // 16 KB: cross-wave partials

    const int hd = blockIdx.x & 7, it = blockIdx.x >> 3;
    const int tid = threadIdx.x, wave = tid >> 5, lane = tid & 31;

    for (int k = tid; k < NN; k += 128) f2s[k] = f2[hd * NN + k];
    for (int k = tid; k < 16 * 128; k += 128)
        msk[k] = amask[(size_t)(it * 16) * 128 + k];
    __syncthreads();

    const int M = lane & 15;                  // query row within tile
    const int kbase = (lane >> 4) * 16;       // f16 A-fragment K base
    const int i = it * 16 + M;
    const float f1i   = f1[hd * NN + i];
    const float mi2e  = rstats[(size_t)(hd * NN + i) * 2] * L2E; // m_i * log2(e)

    const int ncol  = lane & 15;              // B fragment: N column
    const int khalf = (lane >> 4) * 16;       // B fragment: K half
    const _Float16* hTb = hT + (size_t)hd * FOUT * NN + khalf;
    const unsigned int* mrow = msk + M * 128;

    v8f a0 = {}, a1 = {}, a2 = {}, a3 = {};
    const int jstart = wave * 1024;

    for (int j0 = jstart; j0 < jstart + 1024; j0 += 32) {
        const unsigned int w = mrow[j0 >> 5];

        v16h av;
        #pragma unroll
        for (int gq = 0; gq < 2; ++gq)
            #pragma unroll
            for (int e = 0; e < 8; ++e) {
                const int off = kbase + gq * 8 + e;      // j - j0
                float z = f1i + f2s[j0 + off];
                z = fmaxf(z, ALPHA * z);                 // LeakyReLU
                const float p = ((w >> off) & 1u)
                    ? exp2f(fmaf(z, L2E, -mi2e)) : 0.0f; // exp(z - m_i)
                av[gq * 8 + e] = (_Float16)p;            // unnormalized prob
            }

        const _Float16* bp = hTb + j0;
        const v16h b0 = *(const v16h*)(bp + (size_t)(0 * 16 + ncol) * NN);
        const v16h b1 = *(const v16h*)(bp + (size_t)(1 * 16 + ncol) * NN);
        const v16h b2 = *(const v16h*)(bp + (size_t)(2 * 16 + ncol) * NN);
        const v16h b3 = *(const v16h*)(bp + (size_t)(3 * 16 + ncol) * NN);

        a0 = __builtin_amdgcn_wmma_f32_16x16x32_f16(false, av, false, b0, (short)0, a0, false, false);
        a1 = __builtin_amdgcn_wmma_f32_16x16x32_f16(false, av, false, b1, (short)0, a1, false, false);
        a2 = __builtin_amdgcn_wmma_f32_16x16x32_f16(false, av, false, b2, (short)0, a2, false, false);
        a3 = __builtin_amdgcn_wmma_f32_16x16x32_f16(false, av, false, b3, (short)0, a3, false, false);
    }

    #pragma unroll
    for (int r = 0; r < 8; ++r) {
        redo[((wave * 4 + 0) * 8 + r) * 32 + lane] = a0[r];
        redo[((wave * 4 + 1) * 8 + r) * 32 + lane] = a1[r];
        redo[((wave * 4 + 2) * 8 + r) * 32 + lane] = a2[r];
        redo[((wave * 4 + 3) * 8 + r) * 32 + lane] = a3[r];
    }
    __syncthreads();

    // sum the 4 wave-partials, normalize, write out[n][hd*64 + f]
    #pragma unroll
    for (int q = 0; q < 8; ++q) {
        const int idx = tid + q * 128;        // 0..1023
        const int m = idx >> 6, f = idx & 63;
        const int t = f >> 4, n = f & 15;
        const int r = m & 7, ln = n + ((m >> 3) << 4);
        float s = 0.f;
        #pragma unroll
        for (int w2 = 0; w2 < 4; ++w2)
            s += redo[((w2 * 4 + t) * 8 + r) * 32 + ln];
        const float invd = rstats[(size_t)(hd * NN + it * 16 + m) * 2 + 1];
        out[(size_t)(it * 16 + m) * (NH * FOUT) + hd * FOUT + f] = s * invd;
    }
}

// ---------------------------------------------------------------------------
extern "C" void kernel_launch(void* const* d_in, const int* in_sizes, int n_in,
                              void* d_out, int out_size, void* d_ws, size_t ws_size,
                              hipStream_t stream)
{
    const float* x   = (const float*)d_in[0];
    const int*   adj = (const int*)d_in[1];
    const float* W   = (const float*)d_in[2];
    const float* a   = (const float*)d_in[3];
    float* out = (float*)d_out;

    char* ws = (char*)d_ws;
    _Float16*     hT     = (_Float16*)(ws + OFF_HT);
    float*        f1     = (float*)(ws + OFF_F1);
    float*        f2     = (float*)(ws + OFF_F2);
    unsigned int* amask  = (unsigned int*)(ws + OFF_MASK);
    float*        rstats = (float*)(ws + OFF_STATS);

    k1_proj<<<NH * (NN / 16), 128, 0, stream>>>(x, W, a, hT, f1, f2);
    k2_stats<<<NN, 256, 0, stream>>>(adj, f1, f2, amask, rstats);
    k3_attn<<<NH * (NN / 16), 128, 0, stream>>>(hT, f1, f2, amask, rstats, out);
}
